// Head_19902878449989
// MI455X (gfx1250) — matrix operand order repaired and checked by reading, not compiled, not run on hardware
//
#include <hip/hip_runtime.h>
#include <hip/hip_bf16.h>

typedef __attribute__((ext_vector_type(16))) __bf16 v16bf;
typedef __attribute__((ext_vector_type(8)))  __bf16 v8bf;
typedef __attribute__((ext_vector_type(8)))  float  v8f;

#define SZ 512
#define HEADS 64                 // B*T
#define HEAD_ELEMS (SZ * SZ)     // 262144
#define LDP 516                  // padded LDS row pitch (floats)

// ---------------- WMMA fragment loads (per ISA 7.12.2 layouts) ----------------

// A fragment 16x32 bf16: lane (h=lane>>4, m=lane&15) holds
//   elems 0..7  = A[m, k0+8h .. k0+8h+7]
//   elems 8..15 = A[m, k0+16+8h .. k0+16+8h+7]
__device__ __forceinline__ v16bf load_a_frag(const __bf16* __restrict__ A,
                                             int row0, int k0, int lane) {
    const int h = lane >> 4, m = lane & 15;
    const __bf16* p = A + (size_t)(row0 + m) * SZ + k0 + 8 * h;
    v8bf lo = *(const v8bf*)(p);
    v8bf hi = *(const v8bf*)(p + 16);
    v16bf r;
#pragma unroll
    for (int i = 0; i < 8; ++i) { r[i] = lo[i]; r[i + 8] = hi[i]; }
    return r;
}

// B fragment 32x16 bf16 from B stored [N x K] row-major (i.e. computing A*B^T):
// lane (h=lane>>4, n=lane&15) holds elems 0..7 = B[k0+8h+e, n], elems 8..15 = B[k0+16+8h+(e-8), n]
__device__ __forceinline__ v16bf load_b_frag(const __bf16* __restrict__ B,
                                             int col0, int k0, int lane) {
    const int h = lane >> 4, n = lane & 15;
    const __bf16* p = B + (size_t)(col0 + n) * SZ + k0 + 8 * h;
    v8bf lo = *(const v8bf*)(p);
    v8bf hi = *(const v8bf*)(p + 16);
    v16bf r;
#pragma unroll
    for (int i = 0; i < 8; ++i) { r[i] = lo[i]; r[i + 8] = hi[i]; }
    return r;
}

// 32x64 wave tile of C = A * B^T, A:[512x512] bf16 rm, B:[512x512] bf16 rm.
__device__ __forceinline__ void gemm_wave_32x64(const __bf16* __restrict__ A,
                                                const __bf16* __restrict__ B,
                                                int row0, int col0, int lane,
                                                v8f acc[2][4]) {
#pragma unroll 2
    for (int k0 = 0; k0 < SZ; k0 += 32) {
        v16bf a[2], b[4];
#pragma unroll
        for (int mi = 0; mi < 2; ++mi) a[mi] = load_a_frag(A, row0 + 16 * mi, k0, lane);
#pragma unroll
        for (int ni = 0; ni < 4; ++ni) b[ni] = load_b_frag(B, col0 + 16 * ni, k0, lane);
#pragma unroll
        for (int mi = 0; mi < 2; ++mi)
#pragma unroll
            for (int ni = 0; ni < 4; ++ni)
                acc[mi][ni] = __builtin_amdgcn_wmma_f32_16x16x32_bf16(
                    false, a[mi], false, b[ni], (short)0, acc[mi][ni], false, false);
    }
}

// ---------------- elementwise kernels ----------------

// xb = bf16(x + sinusoidal PE); x:[B,S,D] f32 flat
__global__ void pe_convert_kernel(const float* __restrict__ x, __bf16* __restrict__ xb) {
    const int idx = blockIdx.x * 256 + threadIdx.x;  // < B*S*D = 16777216
    const int d = idx & (SZ - 1);
    const int s = (idx >> 9) & 4095;
    const int i2 = d >> 1;  // pair index
    // div = exp(-ln(10000) * (2*i2)/512)
    const float div = __expf(-9.2103403719761836f * (2.0f * (float)i2) * (1.0f / 512.0f));
    const float ang = (float)s * div;
    const float pe = (d & 1) ? __cosf(ang) : __sinf(ang);
    xb[idx] = (__bf16)(x[idx] + pe);
}

__global__ void cvt_bf16_kernel(const float* __restrict__ a, __bf16* __restrict__ o) {
    const int idx = blockIdx.x * 256 + threadIdx.x;
    o[idx] = (__bf16)a[idx];
}

// ---------------- GEMM kernels ----------------

// Out = bf16(Xb * Wb^T + bias); optional transposed store (for V).
// grid = (16, 64), block = 256 (8 waves); wave tile 32x64.
__global__ void proj_kernel(const __bf16* __restrict__ Xb,   // [64][512][512]
                            const __bf16* __restrict__ Wb,   // [8][512][512] (out,in)
                            const float*  __restrict__ bias, // [8][512]
                            __bf16* __restrict__ Out,        // [64][512][512]
                            int transposeStore) {
    const int lane = threadIdx.x & 31;
    const int wave = threadIdx.x >> 5;
    const int head = blockIdx.y;
    const int t = head & 7;
    const int tile = blockIdx.x * 8 + wave;     // 0..127
    const int row0 = (tile >> 3) * 32;          // 16 row tiles
    const int col0 = (tile & 7) * 64;           // 8 col tiles
    const __bf16* A = Xb + (size_t)head * HEAD_ELEMS;
    const __bf16* B = Wb + (size_t)t * HEAD_ELEMS;
    const float*  bs = bias + t * SZ;
    __bf16* O = Out + (size_t)head * HEAD_ELEMS;

    v8f acc[2][4] = {};
    gemm_wave_32x64(A, B, row0, col0, lane, acc);

    const int h = lane >> 4, n = lane & 15;
#pragma unroll
    for (int mi = 0; mi < 2; ++mi)
#pragma unroll
        for (int ni = 0; ni < 4; ++ni)
#pragma unroll
            for (int r = 0; r < 8; ++r) {
                const int m = row0 + mi * 16 + r + 8 * h;
                const int c = col0 + ni * 16 + n;
                const float v = acc[mi][ni][r] + bs[c];
                if (transposeStore)
                    O[(size_t)c * SZ + m] = (__bf16)v;   // store V^T: [d][c]
                else
                    O[(size_t)m * SZ + c] = (__bf16)v;
            }
}

// scores = (K * Q^T) / 512, row softmax, write bf16 attn.
// grid = (16, 64), block = 256; block computes rows [bx*32, +32) x all 512 cols.
__global__ void scores_softmax_kernel(const __bf16* __restrict__ Kb,
                                      const __bf16* __restrict__ Qb,
                                      __bf16* __restrict__ Attn) {
    __shared__ float sc[32 * LDP];  // ~66 KB
    const int lane = threadIdx.x & 31;
    const int wave = threadIdx.x >> 5;
    const int head = blockIdx.y;
    const int row0 = blockIdx.x * 32;
    const int col0 = wave * 64;
    const __bf16* A = Kb + (size_t)head * HEAD_ELEMS;
    const __bf16* B = Qb + (size_t)head * HEAD_ELEMS;

    v8f acc[2][4] = {};
    gemm_wave_32x64(A, B, row0, col0, lane, acc);

    const float inv_d = 1.0f / 512.0f;
    const int h = lane >> 4, n = lane & 15;
#pragma unroll
    for (int mi = 0; mi < 2; ++mi)
#pragma unroll
        for (int ni = 0; ni < 4; ++ni)
#pragma unroll
            for (int r = 0; r < 8; ++r) {
                const int lm = mi * 16 + r + 8 * h;           // 0..31
                const int c  = col0 + ni * 16 + n;            // 0..511
                sc[lm * LDP + c] = acc[mi][ni][r] * inv_d;
            }
    __syncthreads();

    // wave32 row softmax: each wave owns 4 rows
#pragma unroll 1
    for (int i = 0; i < 4; ++i) {
        const int lr = wave * 4 + i;
        float mx = -3.4e38f;
        for (int j = lane; j < SZ; j += 32) mx = fmaxf(mx, sc[lr * LDP + j]);
#pragma unroll
        for (int off = 16; off > 0; off >>= 1) mx = fmaxf(mx, __shfl_xor(mx, off, 32));
        float sum = 0.0f;
        for (int j = lane; j < SZ; j += 32) {
            const float e = __expf(sc[lr * LDP + j] - mx);
            sc[lr * LDP + j] = e;
            sum += e;
        }
#pragma unroll
        for (int off = 16; off > 0; off >>= 1) sum += __shfl_xor(sum, off, 32);
        const float inv = 1.0f / sum;
        __bf16* Arow = Attn + (size_t)head * HEAD_ELEMS + (size_t)(row0 + lr) * SZ;
        for (int j = lane; j < SZ; j += 32) Arow[j] = (__bf16)(sc[lr * LDP + j] * inv);
    }
}

// out = attn * (V^T)^T, f32 store straight into d_out (head-major layout).
__global__ void out_kernel(const __bf16* __restrict__ Attn,  // [64][512][512]
                           const __bf16* __restrict__ Vt,    // [64][512(d)][512(c)]
                           float* __restrict__ Out) {        // [64][512][512]
    const int lane = threadIdx.x & 31;
    const int wave = threadIdx.x >> 5;
    const int head = blockIdx.y;
    const int tile = blockIdx.x * 8 + wave;
    const int row0 = (tile >> 3) * 32;
    const int col0 = (tile & 7) * 64;
    const __bf16* A = Attn + (size_t)head * HEAD_ELEMS;
    const __bf16* B = Vt + (size_t)head * HEAD_ELEMS;
    float* O = Out + (size_t)head * HEAD_ELEMS;

    v8f acc[2][4] = {};
    gemm_wave_32x64(A, B, row0, col0, lane, acc);

    const int h = lane >> 4, n = lane & 15;
#pragma unroll
    for (int mi = 0; mi < 2; ++mi)
#pragma unroll
        for (int ni = 0; ni < 4; ++ni)
#pragma unroll
            for (int r = 0; r < 8; ++r) {
                const int m = row0 + mi * 16 + r + 8 * h;
                const int c = col0 + ni * 16 + n;
                O[(size_t)m * SZ + c] = acc[mi][ni][r];
            }
}

// ---------------- host launch ----------------

extern "C" void kernel_launch(void* const* d_in, const int* in_sizes, int n_in,
                              void* d_out, int out_size, void* d_ws, size_t ws_size,
                              hipStream_t stream) {
    (void)in_sizes; (void)n_in; (void)out_size; (void)ws_size;
    const float* x  = (const float*)d_in[0];
    const float* Wk = (const float*)d_in[1];
    const float* bk = (const float*)d_in[2];
    const float* Wq = (const float*)d_in[3];
    const float* bq = (const float*)d_in[4];
    const float* Wv = (const float*)d_in[5];
    const float* bv = (const float*)d_in[6];
    float* out = (float*)d_out;

    char* ws = (char*)d_ws;
    size_t off = 0;
    __bf16* xb  = (__bf16*)(ws + off); off += (size_t)HEADS * HEAD_ELEMS * 2;  // 32 MB
    __bf16* wkb = (__bf16*)(ws + off); off += (size_t)8 * HEAD_ELEMS * 2;      // 4 MB
    __bf16* wqb = (__bf16*)(ws + off); off += (size_t)8 * HEAD_ELEMS * 2;
    __bf16* wvb = (__bf16*)(ws + off); off += (size_t)8 * HEAD_ELEMS * 2;
    __bf16* kb  = (__bf16*)(ws + off); off += (size_t)HEADS * HEAD_ELEMS * 2;  // 32 MB
    __bf16* qb  = (__bf16*)(ws + off); off += (size_t)HEADS * HEAD_ELEMS * 2;  // 32 MB
    __bf16* vtb = (__bf16*)(ws + off); off += (size_t)HEADS * HEAD_ELEMS * 2;  // 32 MB
    __bf16* attnb = xb;  // xb dead after projections; reuse region for attn

    // 1) PE add + bf16 convert
    pe_convert_kernel<<<(HEADS * HEAD_ELEMS) / 256, 256, 0, stream>>>(x, xb);

    // 2) weight conversion
    cvt_bf16_kernel<<<(8 * HEAD_ELEMS) / 256, 256, 0, stream>>>(Wk, wkb);
    cvt_bf16_kernel<<<(8 * HEAD_ELEMS) / 256, 256, 0, stream>>>(Wq, wqb);
    cvt_bf16_kernel<<<(8 * HEAD_ELEMS) / 256, 256, 0, stream>>>(Wv, wvb);

    // 3) projections (V stored transposed)
    dim3 g(16, HEADS);
    proj_kernel<<<g, 256, 0, stream>>>(xb, wkb, bk, kb, 0);
    proj_kernel<<<g, 256, 0, stream>>>(xb, wqb, bq, qb, 0);
    proj_kernel<<<g, 256, 0, stream>>>(xb, wvb, bv, vtb, 1);

    // 4) scores + fused softmax
    scores_softmax_kernel<<<g, 256, 0, stream>>>(kb, qb, attnb);

    // 5) attn @ V
    out_kernel<<<g, 256, 0, stream>>>(attnb, vtb, out);
}
